// SpatialAwarePixelAttnBlock_59940563583692
// MI455X (gfx1250) — compile-verified, matrix operand order checked by reading, not run on hardware
//
#include <hip/hip_runtime.h>

// ---------------------------------------------------------------------------
// SpatialAwarePixelAttnBlock on MI455X (gfx1250, wave32, WMMA)
//
// Shapes: c=256 channels, N=64*64=4096 pixels.
// Key simplifications (exact w.r.t. the reference math):
//  * top_k masks of M = mvec (x) mvec are column masks given by rank(mvec[j]);
//    nested kk = {2048, 2730, 3072, 3276}  ->  bucket b(j) in {0..4}.
//  * a1*s1+a2*s2+a3*s3+a4*s4 = E[j]*A_b(j) + eps*C_b(j) per row  ->  ONE
//    v @ S_comb GEMM instead of four.
// All GEMMs: bf16 inputs, f32 accumulate via v_wmma_f32_16x16x32_bf16,
// LDS double-buffered tiles, packed b32 LDS scatter for the B^T stage.
// ---------------------------------------------------------------------------

typedef __attribute__((ext_vector_type(16))) __bf16 v16bf;
typedef __attribute__((ext_vector_type(8)))  __bf16 v8bf;
typedef __attribute__((ext_vector_type(2)))  __bf16 v2bf;
typedef __attribute__((ext_vector_type(8)))  float  v8f;

#define CCH 256
#define NPX 4096

// ---------------------------------------------------------------------------
// bf16 weight conversion: Wq,Wk,Wv,Wp (each 256*256) -> contiguous bf16 blocks
// ---------------------------------------------------------------------------
__global__ __launch_bounds__(256) void cvt4_kernel(
    const float* __restrict__ a, const float* __restrict__ b,
    const float* __restrict__ c, const float* __restrict__ d,
    __bf16* __restrict__ out) {
  const int n = CCH * CCH;
  for (int i = blockIdx.x * blockDim.x + threadIdx.x; i < n;
       i += gridDim.x * blockDim.x) {
    out[i]         = (__bf16)a[i];
    out[n + i]     = (__bf16)b[i];
    out[2 * n + i] = (__bf16)c[i];
    out[3 * n + i] = (__bf16)d[i];
  }
}

// ---------------------------------------------------------------------------
// GroupNorm: 32 groups of 8 channels x 4096 px. One block per group.
// ---------------------------------------------------------------------------
__global__ __launch_bounds__(256) void gn_kernel(
    const float* __restrict__ x, const float* __restrict__ gamma,
    const float* __restrict__ beta, __bf16* __restrict__ hn) {
  const int g   = blockIdx.x;
  const int tid = threadIdx.x;
  const int GSZ = 8 * NPX;  // 32768
  const float* xs = x + (size_t)g * GSZ;

  __shared__ float r1[256];
  __shared__ float r2[256];

  float s = 0.f, s2 = 0.f;
  for (int i = tid; i < GSZ; i += 256) {
    float v = xs[i];
    s += v;
    s2 += v * v;
  }
  r1[tid] = s;
  r2[tid] = s2;
  __syncthreads();
  for (int off = 128; off; off >>= 1) {
    if (tid < off) { r1[tid] += r1[tid + off]; r2[tid] += r2[tid + off]; }
    __syncthreads();
  }
  const float mu  = r1[0] * (1.f / GSZ);
  const float var = r2[0] * (1.f / GSZ) - mu * mu;
  const float inv = rsqrtf(var + 1e-6f);

  for (int i = tid; i < GSZ; i += 256) {
    int ch = g * 8 + (i >> 12);
    float v = (xs[i] - mu) * inv * gamma[ch] + beta[ch];
    hn[(size_t)g * GSZ + i] = (__bf16)v;
  }
}

// ---------------------------------------------------------------------------
// Column rank -> bucket. mvec[j] = mask[4*(j/64), 4*(j%64)] (nearest interp).
// rank = #{u: u > v} + #{u == v with smaller index} (matches lax.top_k ties).
// ---------------------------------------------------------------------------
__global__ __launch_bounds__(256) void rank_kernel(
    const float* __restrict__ mask, int* __restrict__ bucket) {
  __shared__ float mv[NPX];
  const int tid = threadIdx.x;
  for (int j = tid; j < NPX; j += 256) {
    int y = j >> 6, x = j & 63;
    mv[j] = mask[(y * 4) * 256 + x * 4];
  }
  __syncthreads();
  const int j0 = blockIdx.x * 256 + tid;
  const float v = mv[j0];
  int rank = 0;
  for (int j = 0; j < NPX; ++j) {
    float u = mv[j];
    rank += (u > v) || (u == v && j < j0);
  }
  int b = 4;
  if (rank < 3276) b = 3;  // (C*4)//5
  if (rank < 3072) b = 2;  // (C*3)//4
  if (rank < 2730) b = 1;  // (C*2)//3
  if (rank < 2048) b = 0;  // C//2
  bucket[j0] = b;
}

// ---------------------------------------------------------------------------
// Fused per-row softmax statistics + combined-S writer.
// One block per row i of w_ (4096 rows).
//   S[i,j] = exp(w-m)*A_{b(j)} + eps*C_{b(j)}   (bf16)
// ---------------------------------------------------------------------------
__global__ __launch_bounds__(256) void softmax_combine_kernel(
    const float* __restrict__ wmat, const int* __restrict__ bucket,
    const float* __restrict__ a1, const float* __restrict__ a2,
    const float* __restrict__ a3, const float* __restrict__ a4,
    __bf16* __restrict__ S) {
  const int i   = blockIdx.x;
  const int tid = threadIdx.x;
  const float* row = wmat + (size_t)i * NPX;

  __shared__ float red[256];
  __shared__ float cA[5];
  __shared__ float cC[5];

  // pass 1: row max
  float m = -1e30f;
  for (int j = tid; j < NPX; j += 256) m = fmaxf(m, row[j]);
  red[tid] = m;
  __syncthreads();
  for (int off = 128; off; off >>= 1) {
    if (tid < off) red[tid] = fmaxf(red[tid], red[tid + off]);
    __syncthreads();
  }
  m = red[0];
  __syncthreads();

  // pass 2: bucketed partial sums P_0..P_3 of exp(w - m)
  float p[4] = {0.f, 0.f, 0.f, 0.f};
  for (int j = tid; j < NPX; j += 256) {
    int b = bucket[j];
    if (b < 4) p[b] += __expf(row[j] - m);
  }
  float P[4];
  for (int b = 0; b < 4; ++b) {
    red[tid] = p[b];
    __syncthreads();
    for (int off = 128; off; off >>= 1) {
      if (tid < off) red[tid] += red[tid + off];
      __syncthreads();
    }
    P[b] = red[0];
    __syncthreads();
  }

  if (tid == 0) {
    const float eps = __expf(-100.0f - m);
    const float av[4]  = {a1[0], a2[0], a3[0], a4[0]};
    const int   cnt[4] = {2048, 1366, 1024, 820};  // N - kk_t
    float Z[4];
    float run = 0.f;
    for (int t = 0; t < 4; ++t) { run += P[t]; Z[t] = run + cnt[t] * eps; }
    cA[4] = 0.f;
    for (int b = 3; b >= 0; --b) cA[b] = cA[b + 1] + av[b] / Z[b];
    float pre = 0.f;
    for (int b = 0; b <= 4; ++b) {
      cC[b] = eps * pre;
      if (b < 4) pre += av[b] / Z[b];
    }
  }
  __syncthreads();

  // pass 3: write combined S row (bf16)
  for (int j = tid; j < NPX; j += 256) {
    int b = bucket[j];
    float E = __expf(row[j] - m);
    S[(size_t)i * NPX + j] = (__bf16)(E * cA[b] + cC[b]);
  }
}

// ---------------------------------------------------------------------------
// bf16 WMMA GEMM: C[m,n] = scale * sum_k A[m,k]*B[k,n]  (+ epilogue)
//   A is MxK row-major, B is KxN row-major.
// Block tile 128x128, K-step 32, 256 threads = 8 waves, wave tile 64x32
// (4 m-tiles x 2 n-tiles of 16x16). LDS is double-buffered (48KB of the
// 320KB WGP pool); next-tile global loads are issued before the fragment
// ds_loads so VMEM latency overlaps the WMMA burst. Fragments are staged in
// the ISA lane swizzle: lane l holds M/N = l%16; halves K = g*8+e and
// 16+g*8+e with g = l/16 -> two aligned 16B ds loads per fragment. The B^T
// stage packs (k,k+1) pairs into b32 LDS stores.
// Epilogues:
//   EPI 0: bf16 store, + bias[m] if bias != null        (k, v, attn-out)
//   EPI 1: f32 w_ store + masked attn1 (bucket[n]==0)   (q^T k)
//   EPI 2: f32 store, + bias[m] + resid[m*N+n]          (proj + residual)
//   EPI 3: bf16 TRANSPOSED store out[n*M+m], + bias[m]  (q -> q^T)
// ---------------------------------------------------------------------------
template <int EPI>
__global__ __launch_bounds__(256) void gemm_bf16_wmma(
    const __bf16* __restrict__ A, const __bf16* __restrict__ B, int M, int N,
    int K, float scale, const float* __restrict__ bias,
    const float* __restrict__ resid, const int* __restrict__ bucket,
    __bf16* __restrict__ outb, float* __restrict__ wstore,
    float* __restrict__ attn_out, float* __restrict__ outf) {
  constexpr int BM = 128, BN = 128, BK = 32;
  constexpr int LDP = 48;  // padded row stride in halves (96B, 16B multiple)

  __shared__ __bf16 As[2][BM * LDP];
  __shared__ __bf16 Bs[2][BN * LDP];  // stored transposed: Bs[n][k]

  const int tid  = threadIdx.x;
  const int lane = tid & 31;
  const int wv   = tid >> 5;
  const int wm   = wv >> 2;      // 0..1
  const int wn   = wv & 3;       // 0..3
  const int g    = lane >> 4;    // half-wave
  const int m16  = lane & 15;
  const int m0   = blockIdx.y * BM;
  const int n0   = blockIdx.x * BN;

  // A stage: 128 rows x 2 segs of 16 halves -> two b128 LDS stores
  const int arow = tid >> 1, aseg = (tid & 1) * 16;
  // B stage: 16 k-pairs x 16 n-segs of 8 -> 8 packed b32 LDS stores
  const int bkp = (tid >> 4) * 2, bns = (tid & 15) * 8;

  auto stage = [&](int buf, int kb) {
    {
      const __bf16* src = A + (size_t)(m0 + arow) * K + kb + aseg;
      v8bf x0 = *(const v8bf*)(src);
      v8bf x1 = *(const v8bf*)(src + 8);
      *(v8bf*)&As[buf][arow * LDP + aseg]     = x0;
      *(v8bf*)&As[buf][arow * LDP + aseg + 8] = x1;
    }
    {
      const __bf16* s0 = B + (size_t)(kb + bkp) * N + n0 + bns;
      v8bf x0 = *(const v8bf*)(s0);
      v8bf x1 = *(const v8bf*)(s0 + N);
#pragma unroll
      for (int e = 0; e < 8; ++e) {
        v2bf t;
        t.x = x0[e];
        t.y = x1[e];
        *(v2bf*)&Bs[buf][(bns + e) * LDP + bkp] = t;
      }
    }
  };

  v8f acc[4][2] = {};

  stage(0, 0);
  __syncthreads();

  int buf = 0;
  for (int k0 = 0; k0 < K; k0 += BK) {
    // issue next tile's global loads / LDS stores first (overlap with WMMA)
    if (k0 + BK < K) {
      stage(buf ^ 1, k0 + BK);
      if (k0 + 2 * BK < K) {  // prefetch tile after next into caches
        __builtin_prefetch(A + (size_t)(m0 + arow) * K + k0 + 2 * BK + aseg, 0, 3);
        __builtin_prefetch(B + (size_t)(k0 + 2 * BK + bkp) * N + n0 + bns, 0, 3);
      }
    }

    // -------- fragments from current buffer --------
    v16bf af[4];
#pragma unroll
    for (int mt = 0; mt < 4; ++mt) {
      const __bf16* p = &As[buf][(wm * 64 + mt * 16 + m16) * LDP + g * 8];
      v8bf lo = *(const v8bf*)(p);
      v8bf hi = *(const v8bf*)(p + 16);
#pragma unroll
      for (int e = 0; e < 8; ++e) { af[mt][e] = lo[e]; af[mt][8 + e] = hi[e]; }
    }
    v16bf bfr[2];
#pragma unroll
    for (int nt = 0; nt < 2; ++nt) {
      const __bf16* p = &Bs[buf][(wn * 32 + nt * 16 + m16) * LDP + g * 8];
      v8bf lo = *(const v8bf*)(p);
      v8bf hi = *(const v8bf*)(p + 16);
#pragma unroll
      for (int e = 0; e < 8; ++e) { bfr[nt][e] = lo[e]; bfr[nt][8 + e] = hi[e]; }
    }

    // -------- 8 WMMAs per K-step --------
#pragma unroll
    for (int mt = 0; mt < 4; ++mt)
#pragma unroll
      for (int nt = 0; nt < 2; ++nt)
        acc[mt][nt] = __builtin_amdgcn_wmma_f32_16x16x32_bf16(
            false, af[mt], false, bfr[nt], (short)0, acc[mt][nt], false, false);

    __syncthreads();
    buf ^= 1;
  }

  // -------- epilogue --------
  // D layout: lane l -> n = l%16 ; vgpr r -> m = r + (l/16)*8
#pragma unroll
  for (int mt = 0; mt < 4; ++mt) {
#pragma unroll
    for (int nt = 0; nt < 2; ++nt) {
      const int mbase = m0 + wm * 64 + mt * 16 + g * 8;
      const int n     = n0 + wn * 32 + nt * 16 + m16;
#pragma unroll
      for (int r = 0; r < 8; ++r) {
        const int m = mbase + r;
        const size_t idx = (size_t)m * N + n;
        float v = acc[mt][nt][r] * scale;
        if (EPI == 0) {
          outb[idx] = (__bf16)(v + (bias ? bias[m] : 0.f));
        } else if (EPI == 1) {
          wstore[idx]   = v;
          attn_out[idx] = (bucket[n] == 0) ? v : -100.0f;
        } else if (EPI == 2) {
          outf[idx] = v + bias[m] + resid[idx];
        } else {  // EPI == 3: transposed bf16 (q -> q^T, row-major N x M)
          outb[(size_t)n * M + m] = (__bf16)(v + bias[m]);
        }
      }
    }
  }
}

// ---------------------------------------------------------------------------
// Host launch
// ---------------------------------------------------------------------------
extern "C" void kernel_launch(void* const* d_in, const int* in_sizes, int n_in,
                              void* d_out, int out_size, void* d_ws,
                              size_t ws_size, hipStream_t stream) {
  (void)in_sizes; (void)n_in; (void)out_size; (void)ws_size;

  const float* x     = (const float*)d_in[0];
  const float* mask  = (const float*)d_in[1];
  const float* gamma = (const float*)d_in[2];
  const float* beta  = (const float*)d_in[3];
  const float* Wq    = (const float*)d_in[4];
  const float* bq    = (const float*)d_in[5];
  const float* Wk    = (const float*)d_in[6];
  const float* bk    = (const float*)d_in[7];
  const float* Wv    = (const float*)d_in[8];
  const float* bv    = (const float*)d_in[9];
  const float* Wp    = (const float*)d_in[10];
  const float* bp    = (const float*)d_in[11];
  const float* a1    = (const float*)d_in[12];
  const float* a2    = (const float*)d_in[13];
  const float* a3    = (const float*)d_in[14];
  const float* a4    = (const float*)d_in[15];

  float* out_y    = (float*)d_out;                      // (256,64,64)
  float* out_attn = (float*)d_out + (size_t)CCH * NPX;  // (4096,4096)

  // workspace carve-out (256B aligned)
  char* base = (char*)d_ws;
  size_t off = 0;
  auto take = [&](size_t bytes) -> char* {
    char* p = base + off;
    off = (off + bytes + 255) & ~(size_t)255;
    return p;
  };
  __bf16* hn   = (__bf16*)take((size_t)CCH * NPX * 2);
  __bf16* Wb   = (__bf16*)take((size_t)4 * CCH * CCH * 2);  // Wq,Wk,Wv,Wp
  __bf16* qT   = (__bf16*)take((size_t)CCH * NPX * 2);      // (4096 x 256)
  __bf16* kb   = (__bf16*)take((size_t)CCH * NPX * 2);
  __bf16* vb   = (__bf16*)take((size_t)CCH * NPX * 2);
  int*    bkt  = (int*)take((size_t)NPX * 4);
  float*  wmat = (float*)take((size_t)NPX * NPX * 4);
  __bf16* S    = (__bf16*)take((size_t)NPX * NPX * 2);
  __bf16* outb = (__bf16*)take((size_t)CCH * NPX * 2);

  const __bf16* WqB = Wb;
  const __bf16* WkB = Wb + 1 * CCH * CCH;
  const __bf16* WvB = Wb + 2 * CCH * CCH;
  const __bf16* WpB = Wb + 3 * CCH * CCH;

  // 1) convert weights to bf16
  cvt4_kernel<<<256, 256, 0, stream>>>(Wq, Wk, Wv, Wp, Wb);

  // 2) GroupNorm -> hn (bf16)
  gn_kernel<<<32, 256, 0, stream>>>(x, gamma, beta, hn);

  // 3) QKV: (256x256) @ (256x4096) + bias ; q stored transposed
  dim3 gqkv(NPX / 128, CCH / 128);
  gemm_bf16_wmma<3><<<gqkv, 256, 0, stream>>>(
      WqB, hn, CCH, NPX, CCH, 1.f, bq, nullptr, nullptr, qT, nullptr, nullptr,
      nullptr);
  gemm_bf16_wmma<0><<<gqkv, 256, 0, stream>>>(
      WkB, hn, CCH, NPX, CCH, 1.f, bk, nullptr, nullptr, kb, nullptr, nullptr,
      nullptr);
  gemm_bf16_wmma<0><<<gqkv, 256, 0, stream>>>(
      WvB, hn, CCH, NPX, CCH, 1.f, bv, nullptr, nullptr, vb, nullptr, nullptr,
      nullptr);

  // 4) column ranks -> buckets
  rank_kernel<<<16, 256, 0, stream>>>(mask, bkt);

  // 5) w_ = q^T k / 16 ; also emit attn1 (h_sub) masked with bucket==0
  dim3 gw(NPX / 128, NPX / 128);
  gemm_bf16_wmma<1><<<gw, 256, 0, stream>>>(
      qT, kb, NPX, NPX, CCH, 0.0625f, nullptr, nullptr, bkt, nullptr, wmat,
      out_attn, nullptr);

  // 6) per-row softmax stats + combined S (bf16)
  softmax_combine_kernel<<<NPX, 256, 0, stream>>>(wmat, bkt, a1, a2, a3, a4, S);

  // 7) out = v @ S   (256 x 4096 x 4096) -> bf16
  gemm_bf16_wmma<0><<<gqkv, 256, 0, stream>>>(
      vb, S, CCH, NPX, NPX, 1.f, nullptr, nullptr, nullptr, outb, nullptr,
      nullptr, nullptr);

  // 8) y = Wp @ out + bp + x  -> f32 d_out
  gemm_bf16_wmma<2><<<gqkv, 256, 0, stream>>>(
      WpB, outb, CCH, NPX, CCH, 1.f, bp, x, nullptr, nullptr, nullptr, nullptr,
      out_y);
}
